// MultiHeadAttention_73675868995934
// MI455X (gfx1250) — matrix operand order repaired and checked by reading, not compiled, not run on hardware
//
#include <hip/hip_runtime.h>
#include <hip/hip_bf16.h>

// ---------------------------------------------------------------------------
// MI455X (gfx1250) fused MHA block. wave32, WMMA bf16 (f32 accum).
// Memory-bound problem (~1.3 GB min traffic @ 23.3 TB/s): attn matrix is
// written once (fused scores+mask+softmax via 128KB LDS strip) and read once
// (context GEMM). All GEMMs use v_wmma_f32_16x16x32_bf16.
// ---------------------------------------------------------------------------

typedef __attribute__((ext_vector_type(16))) __bf16 bf16x16;
typedef __attribute__((ext_vector_type(8)))  __bf16 bf16x8;
typedef __attribute__((ext_vector_type(4)))  __bf16 bf16x4;
typedef __attribute__((ext_vector_type(8)))  float  f32x8;
typedef __attribute__((ext_vector_type(4)))  float  f32x4;

constexpr int kB  = 4;
constexpr int kS  = 2048;
constexpr int kDM = 512;
constexpr int kH  = 8;
constexpr int kDH = 64;
constexpr int kNTOK  = kB * kS;   // 8192 rows
constexpr int kHEADS = kB * kH;   // 32 (b,h) head chunks, each contiguous [2048 x 64]

// ---- WMMA helpers (ISA 7.12.2 layouts, wave32) ------------------------------

__device__ __forceinline__ f32x8 wmma_bf16(bf16x16 a, bf16x16 b, f32x8 c) {
  // (neg_a, A, neg_b, B, c_mod, C, reuse_a, reuse_b)
  return __builtin_amdgcn_wmma_f32_16x16x32_bf16(false, a, false, b, (short)0, c,
                                                 false, false);
}

// 16x32 bf16 A-tile from row-major [.. x ld] at (row0, k0).
// lane: m = lane&15, hi = lane>>4; comps 0..7 = K(k0+8*hi..), 8..15 = +16.
__device__ __forceinline__ bf16x16 load_a16(const __bf16* base, int ld, int row0, int k0) {
  int lane = threadIdx.x & 31;
  int m = lane & 15, hi = lane >> 4;
  const __bf16* p = base + (size_t)(row0 + m) * ld + k0 + 8 * hi;
  union { bf16x16 v; bf16x8 h[2]; } u;
  u.h[0] = *(const bf16x8*)p;
  u.h[1] = *(const bf16x8*)(p + 16);
  return u.v;
}

// Same A-tile but from an fp32 row-major source, converting to bf16 inline.
__device__ __forceinline__ bf16x16 load_a16_f32(const float* base, int ld, int row0, int k0) {
  int lane = threadIdx.x & 31;
  int m = lane & 15, hi = lane >> 4;
  const float* p = base + (size_t)(row0 + m) * ld + k0 + 8 * hi;
  f32x4 a0 = *(const f32x4*)(p);
  f32x4 a1 = *(const f32x4*)(p + 4);
  f32x4 a2 = *(const f32x4*)(p + 16);
  f32x4 a3 = *(const f32x4*)(p + 20);
  bf16x16 r;
#pragma unroll
  for (int i = 0; i < 4; ++i) {
    r[i]      = (__bf16)a0[i];
    r[4 + i]  = (__bf16)a1[i];
    r[8 + i]  = (__bf16)a2[i];
    r[12 + i] = (__bf16)a3[i];
  }
  return r;
}

// 32x16 bf16 B-tile where the source is "N-major": src row (n0+n) supplies
// B column n, i.e. B[k][n] = src[(n0+n)*ld + k]. Covers X@W^T (W row-major)
// and Q@K^T (K rows) and attn@V via V^T. Comps 0..7 = B[k0+8*hi + j][n], 8..15 = +16.
__device__ __forceinline__ bf16x16 load_bT16(const __bf16* base, int ld, int n0, int k0) {
  int lane = threadIdx.x & 31;
  int n = lane & 15, hi = lane >> 4;
  const __bf16* p = base + (size_t)(n0 + n) * ld + k0 + 8 * hi;
  union { bf16x16 v; bf16x8 h[2]; } u;
  u.h[0] = *(const bf16x8*)p;
  u.h[1] = *(const bf16x8*)(p + 16);
  return u.v;
}

// ---- Kernel 0: fp32 -> bf16 convert (vec4) ---------------------------------

__global__ void cvt_f32_bf16(const float* __restrict__ src, __bf16* __restrict__ dst, int n4) {
  int i = blockIdx.x * blockDim.x + threadIdx.x;
  if (i >= n4) return;
  f32x4 v = ((const f32x4*)src)[i];
  bf16x4 o;
#pragma unroll
  for (int c = 0; c < 4; ++c) o[c] = (__bf16)v[c];
  ((bf16x4*)dst)[i] = o;
}

// ---- Kernel 1: QKV projections. Q=Xq@WQ^T, K=Xk@WK^T, V=Xv@WK^T (bug kept).
// One wave computes a 16x64 tile (A-frag reused across 4 N-tiles). ----------

__global__ void qkv_gemm(const __bf16* __restrict__ Xq, const __bf16* __restrict__ Xk,
                         const __bf16* __restrict__ Xv, const __bf16* __restrict__ WQ,
                         const __bf16* __restrict__ WK, __bf16* __restrict__ Qo,
                         __bf16* __restrict__ Ko, __bf16* __restrict__ Vo) {
  int wave = (blockIdx.x * blockDim.x + threadIdx.x) >> 5;
  int out  = wave / 4096;            // 0:Q 1:K 2:V
  int t    = wave % 4096;            // 512 row-tiles x 8 col-groups
  int row0 = (t >> 3) * 16;
  int n0   = (t & 7) * 64;
  const __bf16* X = (out == 0) ? Xq : ((out == 1) ? Xk : Xv);
  const __bf16* W = (out == 0) ? WQ : WK;     // V uses W_K, as in the source
  __bf16* C = (out == 0) ? Qo : ((out == 1) ? Ko : Vo);

  f32x8 acc[4] = {};
  for (int k0 = 0; k0 < kDM; k0 += 32) {
    bf16x16 a = load_a16(X, kDM, row0, k0);
#pragma unroll
    for (int j = 0; j < 4; ++j) {
      bf16x16 b = load_bT16(W, kDM, n0 + 16 * j, k0);
      acc[j] = wmma_bf16(a, b, acc[j]);
    }
  }
  int lane = threadIdx.x & 31;
  int n = lane & 15, hi = lane >> 4;
#pragma unroll
  for (int j = 0; j < 4; ++j)
#pragma unroll
    for (int i = 0; i < 8; ++i)
      C[(size_t)(row0 + i + 8 * hi) * kDM + n0 + 16 * j + n] = (__bf16)acc[j][i];
}

// ---- Kernel 1b: per-head V transpose: V[head][s][e] -> VT[head][e][s] ------

__global__ void vtranspose(const __bf16* __restrict__ V, __bf16* __restrict__ VT) {
  __shared__ __bf16 tile[64][65];
  int blk = blockIdx.x;                 // HEADS * (S/64) = 32*32 = 1024
  int s0 = (blk & 31) * 64;
  int bh = blk >> 5;
  const __bf16* Vh  = V  + (size_t)bh * (kS * kDH);   // contiguous [2048 x 64]
  __bf16*       VTh = VT + (size_t)bh * (kDH * kS);   // [64 x 2048]
  int t = threadIdx.x;
#pragma unroll
  for (int i = 0; i < 16; ++i) {
    int idx = t + i * 256, r = idx >> 6, c = idx & 63;
    tile[r][c] = Vh[(size_t)(s0 + r) * kDH + c];
  }
  __syncthreads();
#pragma unroll
  for (int i = 0; i < 16; ++i) {
    int idx = t + i * 256, e = idx >> 6, c = idx & 63;
    VTh[(size_t)e * kS + s0 + c] = tile[c][e];
  }
}

// ---- Kernel 2: fused scores + mask + softmax. One block = 16 q-rows of one
// head; 4 waves fill a 16x2048 fp32 score strip in LDS (128KB), then do the
// row softmax and write normalized attn to d_out exactly once. --------------

__global__ void attn_softmax(const __bf16* __restrict__ Q, const __bf16* __restrict__ K,
                             const unsigned char* __restrict__ mask,
                             float* __restrict__ attn) {
  extern __shared__ float sc[];            // 16 * 2048 floats
  int blk = blockIdx.x;                    // HEADS * (S/16) = 32*128
  int qt = blk & 127, bh = blk >> 7, b = bh >> 3;
  int q0 = qt * 16;
  const __bf16* Qh = Q + (size_t)bh * (kS * kDH);
  const __bf16* Kh = K + (size_t)bh * (kS * kDH);
  int wave = threadIdx.x >> 5;
  int lane = threadIdx.x & 31;
  int n = lane & 15, hi = lane >> 4;

  // Phase 1: scores. A-frags (Q 16x64) loaded once; each wave covers 512 cols.
  bf16x16 a0 = load_a16(Qh, kDH, q0, 0);
  bf16x16 a1 = load_a16(Qh, kDH, q0, 32);
  const unsigned char* mrow = mask + (size_t)b * kS * kS;
  for (int nt = wave * 32; nt < wave * 32 + 32; ++nt) {
    int n0 = nt * 16;
    f32x8 c = {};
    c = wmma_bf16(a0, load_bT16(Kh, kDH, n0, 0), c);
    c = wmma_bf16(a1, load_bT16(Kh, kDH, n0, 32), c);
#pragma unroll
    for (int i = 0; i < 8; ++i) {
      int m = i + 8 * hi, col = n0 + n;
      unsigned char msk = mrow[(size_t)(q0 + m) * kS + col];
      sc[m * kS + col] = msk ? -1e9f : c[i] * 0.125f;   // 1/sqrt(64)
    }
  }
  __syncthreads();

  // Phase 2: row softmax; each wave owns 4 rows; vec4 lane-strided.
  for (int r = wave * 4; r < wave * 4 + 4; ++r) {
    f32x4* rv = (f32x4*)(sc + r * kS);
    float mx = -3.4e38f;
    for (int j = lane; j < kS / 4; j += 32) {
      f32x4 v = rv[j];
      mx = fmaxf(mx, fmaxf(fmaxf(v[0], v[1]), fmaxf(v[2], v[3])));
    }
#pragma unroll
    for (int o = 16; o > 0; o >>= 1) mx = fmaxf(mx, __shfl_xor(mx, o, 32));
    float sum = 0.f;
    for (int j = lane; j < kS / 4; j += 32) {
      f32x4 v = rv[j];
#pragma unroll
      for (int cth = 0; cth < 4; ++cth) { v[cth] = __expf(v[cth] - mx); sum += v[cth]; }
      rv[j] = v;
    }
#pragma unroll
    for (int o = 16; o > 0; o >>= 1) sum += __shfl_xor(sum, o, 32);
    float inv = 1.f / sum;
    f32x4* ov = (f32x4*)(attn + (size_t)bh * kS * kS + (size_t)(q0 + r) * kS);
    for (int j = lane; j < kS / 4; j += 32) { f32x4 v = rv[j]; v *= inv; ov[j] = v; }
  }
}

// ---- Kernel 3: context = attn @ V per head (A fp32 -> bf16 inline, B = V^T).
// Scatter into ctx[b][s][h*64+e] (the transpose(0,2,1,3) layout) as bf16. ---

__global__ void context_gemm(const float* __restrict__ attn, const __bf16* __restrict__ VT,
                             __bf16* __restrict__ ctx) {
  int wave = (blockIdx.x * blockDim.x + threadIdx.x) >> 5;   // 4096 waves
  int bh = wave >> 7;
  int q0 = (wave & 127) * 16;
  const float*  A  = attn + (size_t)bh * kS * kS;
  const __bf16* Bm = VT + (size_t)bh * (kDH * kS);           // [64 x 2048]
  f32x8 acc[4] = {};
  for (int k0 = 0; k0 < kS; k0 += 32) {
    bf16x16 a = load_a16_f32(A, kS, q0, k0);
#pragma unroll
    for (int j = 0; j < 4; ++j) {
      bf16x16 b = load_bT16(Bm, kS, 16 * j, k0);
      acc[j] = wmma_bf16(a, b, acc[j]);
    }
  }
  int lane = threadIdx.x & 31;
  int n = lane & 15, hi = lane >> 4;
  int b = bh >> 3, h = bh & 7;
  __bf16* C = ctx + (size_t)b * kS * kDM + (size_t)h * kDH;
#pragma unroll
  for (int j = 0; j < 4; ++j)
#pragma unroll
    for (int i = 0; i < 8; ++i)
      C[(size_t)(q0 + i + 8 * hi) * kDM + 16 * j + n] = (__bf16)acc[j][i];
}

// ---- Kernel 4: y = ctx @ Wfc^T + input_Q (residual), fp32 out -------------

__global__ void fc_residual(const __bf16* __restrict__ ctx, const __bf16* __restrict__ Wfc,
                            const float* __restrict__ Xq, float* __restrict__ y) {
  int wave = (blockIdx.x * blockDim.x + threadIdx.x) >> 5;   // 4096 waves
  int row0 = (wave >> 3) * 16;
  int n0   = (wave & 7) * 64;
  f32x8 acc[4] = {};
  for (int k0 = 0; k0 < kDM; k0 += 32) {
    bf16x16 a = load_a16(ctx, kDM, row0, k0);
#pragma unroll
    for (int j = 0; j < 4; ++j) {
      bf16x16 b = load_bT16(Wfc, kDM, n0 + 16 * j, k0);
      acc[j] = wmma_bf16(a, b, acc[j]);
    }
  }
  int lane = threadIdx.x & 31;
  int n = lane & 15, hi = lane >> 4;
#pragma unroll
  for (int j = 0; j < 4; ++j)
#pragma unroll
    for (int i = 0; i < 8; ++i) {
      size_t idx = (size_t)(row0 + i + 8 * hi) * kDM + n0 + 16 * j + n;
      y[idx] = acc[j][i] + Xq[idx];
    }
}

// ---- Kernel 5: LayerNorm per 512-wide row, one wave per row ---------------

__global__ void layernorm(const float* __restrict__ y, const float* __restrict__ gamma,
                          const float* __restrict__ beta, float* __restrict__ out) {
  int row = (blockIdx.x * blockDim.x + threadIdx.x) >> 5;    // 8192 rows
  int lane = threadIdx.x & 31;
  const f32x4* rv = (const f32x4*)(y + (size_t)row * kDM);
  f32x4 vals[4];
  float s = 0.f;
#pragma unroll
  for (int j = 0; j < 4; ++j) {
    vals[j] = rv[lane + j * 32];
#pragma unroll
    for (int c = 0; c < 4; ++c) s += vals[j][c];
  }
#pragma unroll
  for (int o = 16; o > 0; o >>= 1) s += __shfl_xor(s, o, 32);
  float mean = s * (1.f / kDM);
  float vs = 0.f;
#pragma unroll
  for (int j = 0; j < 4; ++j)
#pragma unroll
    for (int c = 0; c < 4; ++c) { float d = vals[j][c] - mean; vs += d * d; }
#pragma unroll
  for (int o = 16; o > 0; o >>= 1) vs += __shfl_xor(vs, o, 32);
  float inv = rsqrtf(vs * (1.f / kDM) + 1e-5f);
  const f32x4* gv = (const f32x4*)gamma;
  const f32x4* bv = (const f32x4*)beta;
  f32x4* ov = (f32x4*)(out + (size_t)row * kDM);
#pragma unroll
  for (int j = 0; j < 4; ++j) {
    f32x4 g = gv[lane + j * 32], bb = bv[lane + j * 32], v = vals[j], o4;
#pragma unroll
    for (int c = 0; c < 4; ++c) o4[c] = (v[c] - mean) * inv * g[c] + bb[c];
    ov[lane + j * 32] = o4;
  }
}

// ---------------------------------------------------------------------------

extern "C" void kernel_launch(void* const* d_in, const int* in_sizes, int n_in,
                              void* d_out, int out_size, void* d_ws, size_t ws_size,
                              hipStream_t stream) {
  (void)in_sizes; (void)n_in; (void)out_size; (void)ws_size;
  const float* Xq   = (const float*)d_in[0];
  const float* Xk   = (const float*)d_in[1];
  const float* Xv   = (const float*)d_in[2];
  const unsigned char* mask = (const unsigned char*)d_in[3];  // jax bool = 1 byte
  const float* WQ   = (const float*)d_in[4];
  const float* WK   = (const float*)d_in[5];
  // d_in[6] = W_V intentionally unused (reference bug: V projected with W_K)
  const float* Wfc  = (const float*)d_in[7];
  const float* gamma = (const float*)d_in[8];
  const float* beta  = (const float*)d_in[9];

  float* normed_out = (float*)d_out;
  float* attn_out   = normed_out + (size_t)kNTOK * kDM;       // 4M floats in

  char* w = (char*)d_ws;
  auto take = [&](size_t bytes) -> void* {
    void* p = (void*)w; w += (bytes + 255) & ~(size_t)255; return p;
  };
  const size_t tok_bf = (size_t)kNTOK * kDM * sizeof(__bf16); // 8 MB
  __bf16* Xq_bf  = (__bf16*)take(tok_bf);
  __bf16* Xk_bf  = (__bf16*)take(tok_bf);
  __bf16* Xv_bf  = (__bf16*)take(tok_bf);
  __bf16* WQ_bf  = (__bf16*)take((size_t)kDM * kDM * 2);
  __bf16* WK_bf  = (__bf16*)take((size_t)kDM * kDM * 2);
  __bf16* Wfc_bf = (__bf16*)take((size_t)kDM * kDM * 2);
  __bf16* Qbf    = (__bf16*)take(tok_bf);
  __bf16* Kbf    = (__bf16*)take(tok_bf);
  __bf16* Vbf    = (__bf16*)take(tok_bf);
  __bf16* VT     = (__bf16*)take(tok_bf);
  __bf16* ctxbf  = (__bf16*)take(tok_bf);
  float*  ybuf   = (float*)take((size_t)kNTOK * kDM * sizeof(float));

  // 0) fp32 -> bf16 converts
  cvt_f32_bf16<<<(kNTOK * kDM / 4 + 255) / 256, 256, 0, stream>>>(Xq, Xq_bf, kNTOK * kDM / 4);
  cvt_f32_bf16<<<(kNTOK * kDM / 4 + 255) / 256, 256, 0, stream>>>(Xk, Xk_bf, kNTOK * kDM / 4);
  cvt_f32_bf16<<<(kNTOK * kDM / 4 + 255) / 256, 256, 0, stream>>>(Xv, Xv_bf, kNTOK * kDM / 4);
  cvt_f32_bf16<<<(kDM * kDM / 4 + 255) / 256, 256, 0, stream>>>(WQ, WQ_bf, kDM * kDM / 4);
  cvt_f32_bf16<<<(kDM * kDM / 4 + 255) / 256, 256, 0, stream>>>(WK, WK_bf, kDM * kDM / 4);
  cvt_f32_bf16<<<(kDM * kDM / 4 + 255) / 256, 256, 0, stream>>>(Wfc, Wfc_bf, kDM * kDM / 4);

  // 1) QKV projections: 3 * 4096 waves, 4 waves/block
  qkv_gemm<<<3072, 128, 0, stream>>>(Xq_bf, Xk_bf, Xv_bf, WQ_bf, WK_bf, Qbf, Kbf, Vbf);

  // 1b) per-head V transpose
  vtranspose<<<kHEADS * (kS / 64), 256, 0, stream>>>(Vbf, VT);

  // 2) fused scores+mask+softmax: one block per (head, 16 q-rows); 128KB LDS
  attn_softmax<<<kHEADS * (kS / 16), 128, 16 * kS * sizeof(float), stream>>>(
      Qbf, Kbf, mask, attn_out);

  // 3) context = attn @ V  (4096 waves)
  context_gemm<<<1024, 128, 0, stream>>>(attn_out, VT, ctxbf);

  // 4) output projection + residual (4096 waves)
  fc_residual<<<1024, 128, 0, stream>>>(ctxbf, Wfc_bf, Xq, ybuf);

  // 5) LayerNorm -> normed output (one wave per row)
  layernorm<<<kNTOK / 8, 256, 0, stream>>>(ybuf, gamma, beta, normed_out);
}
